// GlobalAttentionPooling_47588237639683
// MI455X (gfx1250) — matrix operand order repaired and checked by reading, not compiled, not run on hardware
//
#include <hip/hip_runtime.h>
#include <stdint.h>

#define HIDDEN 1024
#define HALF   512
#define NGRAPH 256
#define NTILES (HALF / 16)    // 32 n-tiles of h columns
#define KTILES (HIDDEN / 32)  // 32 k-tiles of depth 32

typedef __attribute__((ext_vector_type(16))) __bf16   bf16x16;
typedef __attribute__((ext_vector_type(8)))  float    floatx8;
typedef __attribute__((ext_vector_type(4)))  float    floatx4;
typedef __attribute__((ext_vector_type(4)))  unsigned uintx4;
typedef __attribute__((ext_vector_type(2)))  unsigned uintx2;

union Frag16 { bf16x16 v; uintx4 q[2]; };

__device__ __forceinline__ unsigned short f2bf(float x) {
    unsigned u = __float_as_uint(x);
    unsigned r = u + 0x7FFFu + ((u >> 16) & 1u);   // round-to-nearest-even
    return (unsigned short)(r >> 16);
}

__device__ __forceinline__ unsigned f2key(float x) {
    unsigned u = __float_as_uint(x);
    return (u & 0x80000000u) ? ~u : (u | 0x80000000u);  // order-preserving key
}
__device__ __forceinline__ float key2f(unsigned k) {
    unsigned u = (k & 0x80000000u) ? (k & 0x7FFFFFFFu) : ~k;
    return __uint_as_float(u);
}

// ---------------------------------------------------------------- init
__global__ void init_kernel(float* __restrict__ out, int outN,
                            unsigned* __restrict__ maxkey, float* __restrict__ sumv) {
    int i = blockIdx.x * blockDim.x + threadIdx.x;
    if (i < outN) out[i] = 0.0f;
    if (i == 0) { *maxkey = 0u; *sumv = 0.0f; }
}

// ------------------------------------------------- repack W1 -> bf16 B-fragments
// Consumption: fragment f = nt*32 + kt; lane l reads 8 dwords at W1p[(f*32 + l)*8 + v].
// B order used here (contiguous K, same as A): VGPR v packs K={2v,2v+1}; lanes 0-15 hold K 0..15 (N=lane),
// lanes 16-31 hold K 16..31 (N=lane-16).
__global__ void repack_w1(const float* __restrict__ W1, uint32_t* __restrict__ W1p) {
    int idx = blockIdx.x * blockDim.x + threadIdx.x;
    if (idx >= HIDDEN * HALF / 2) return;
    int v    = idx & 7;
    int lane = (idx >> 3) & 31;
    int kt   = (idx >> 8) & 31;
    int nt   = idx >> 13;
    int n = nt * 16 + (lane & 15);
    int k = kt * 32 + ((lane >> 4) * 16) + 2 * v;
    unsigned short lo = f2bf(W1[(size_t)k * HALF + n]);
    unsigned short hi = f2bf(W1[(size_t)(k + 1) * HALF + n]);
    W1p[idx] = (uint32_t)lo | ((uint32_t)hi << 16);
}

// ------------------------------------------------- fused scores = (relu(X@W1+b1))@W2 + b2
__global__ __launch_bounds__(64)
void scores_kernel(const float* __restrict__ X, const uint32_t* __restrict__ W1p,
                   const float* __restrict__ b1, const float* __restrict__ W2,
                   const float* __restrict__ b2, float* __restrict__ scores, int N) {
    __shared__ __align__(16) unsigned short ldsA[2][16 * HIDDEN];  // 64 KB
    const int w    = threadIdx.x >> 5;
    const int lane = threadIdx.x & 31;
    const long rowBase = ((long)blockIdx.x * 2 + w) * 16;
    if (rowBase >= N) return;  // uniform per wave

    // Stage 16x1024 f32 row block -> bf16 in LDS (row-major)
    const float* xblk = X + rowBase * HIDDEN;
    unsigned short* As = &ldsA[w][0];
    for (int i = lane * 4; i < 16 * HIDDEN; i += 32 * 4) {
        floatx4 f = *(const floatx4*)(xblk + i);
        uintx2 p;
        p.x = (uint32_t)f2bf(f.x) | ((uint32_t)f2bf(f.y) << 16);
        p.y = (uint32_t)f2bf(f.z) | ((uint32_t)f2bf(f.w) << 16);
        *(uintx2*)(As + i) = p;
    }

    const int nlo = lane & 15;   // N (and A row) within tile
    const int hi  = lane >> 4;   // which K half this lane carries for A
    // A order used here (contiguous K, same as B): lanes 0-15: q0=K0..7, q1=K8..15; lanes 16-31: q0=K16..23, q1=K24..31
    const unsigned short* arow = As + nlo * HIDDEN + hi * 16;

    float sacc[8] = {0.f, 0.f, 0.f, 0.f, 0.f, 0.f, 0.f, 0.f};

    for (int nt = 0; nt < NTILES; ++nt) {
        floatx8 acc = {0.f, 0.f, 0.f, 0.f, 0.f, 0.f, 0.f, 0.f};
        const uintx4* bp = ((const uintx4*)W1p) + (((size_t)nt * 32) * 32 + lane) * 2;
#pragma unroll 4
        for (int kt = 0; kt < KTILES; ++kt) {
            Frag16 af, bf;
            af.q[0] = *(const uintx4*)(arow + kt * 32);
            af.q[1] = *(const uintx4*)(arow + kt * 32 + 8);
            bf.q[0] = bp[(size_t)kt * 64];
            bf.q[1] = bp[(size_t)kt * 64 + 1];
            acc = __builtin_amdgcn_wmma_f32_16x16x32_bf16(
                false, af.v, false, bf.v, (short)0, acc, false, false);
        }
        const float b1v = b1[nt * 16 + nlo];
        const float w2v = W2[nt * 16 + nlo];
#pragma unroll
        for (int j = 0; j < 8; ++j) {
            float h = acc[j] + b1v;
            h = fmaxf(h, 0.0f);
            sacc[j] = fmaf(h, w2v, sacc[j]);
        }
    }

    const float b2v = b2[0];
#pragma unroll
    for (int j = 0; j < 8; ++j) {
        float r = sacc[j];
        r += __shfl_xor(r, 1, 16);
        r += __shfl_xor(r, 2, 16);
        r += __shfl_xor(r, 4, 16);
        r += __shfl_xor(r, 8, 16);
        if (nlo == 0) scores[rowBase + j + hi * 8] = r + b2v;  // C: lanes0-15 -> M=j, lanes16-31 -> M=8+j
    }
}

// ---------------------------------------------------------------- softmax pieces
__global__ void reduce_max(const float* __restrict__ s, int N, unsigned* __restrict__ maxkey) {
    __shared__ unsigned sm[256];
    unsigned best = 0u;
    for (int i = blockIdx.x * blockDim.x + threadIdx.x; i < N; i += gridDim.x * blockDim.x)
        best = max(best, f2key(s[i]));
    sm[threadIdx.x] = best;
    __syncthreads();
    for (int o = 128; o > 0; o >>= 1) {
        if (threadIdx.x < o) sm[threadIdx.x] = max(sm[threadIdx.x], sm[threadIdx.x + o]);
        __syncthreads();
    }
    if (threadIdx.x == 0) atomicMax(maxkey, sm[0]);
}

__global__ void reduce_sumexp(const float* __restrict__ s, int N,
                              const unsigned* __restrict__ maxkey, float* __restrict__ sumv) {
    __shared__ float sm[256];
    const float m = key2f(*maxkey);
    float acc = 0.0f;
    for (int i = blockIdx.x * blockDim.x + threadIdx.x; i < N; i += gridDim.x * blockDim.x)
        acc += expf(s[i] - m);
    sm[threadIdx.x] = acc;
    __syncthreads();
    for (int o = 128; o > 0; o >>= 1) {
        if (threadIdx.x < o) sm[threadIdx.x] += sm[threadIdx.x + o];
        __syncthreads();
    }
    if (threadIdx.x == 0) atomicAdd(sumv, sm[0]);
}

__global__ void normalize_weights(float* __restrict__ s, int N,
                                  const unsigned* __restrict__ maxkey,
                                  const float* __restrict__ sumv) {
    int i = blockIdx.x * blockDim.x + threadIdx.x;
    if (i >= N) return;
    const float m = key2f(*maxkey);
    const float inv = 1.0f / *sumv;
    s[i] = expf(s[i] - m) * inv;
}

// ---------------------------------------------------------------- weighted segment-sum
// grid = 16 column-chunks x 32 row-segments; LDS tile [256 graphs][64 cols]
__global__ __launch_bounds__(256)
void scatter_kernel(const float* __restrict__ X, const float* __restrict__ wts,
                    const int* __restrict__ bidx, float* __restrict__ out,
                    int N, int numSeg) {
    __shared__ float acc[NGRAPH * 64];  // 64 KB
    for (int i = threadIdx.x; i < NGRAPH * 64; i += 256) acc[i] = 0.0f;
    __syncthreads();

    const int colChunk = blockIdx.x & 15;
    const int seg      = blockIdx.x >> 4;
    const int c0       = colChunk * 64;
    const int segLen   = (N + numSeg - 1) / numSeg;
    const int r0 = seg * segLen;
    const int r1 = min(N, r0 + segLen);
    const int c  = threadIdx.x & 63;

    for (int i = r0 + (threadIdx.x >> 6); i < r1; i += 4) {
        const float wv = wts[i];
        const int   g  = bidx[i];
        const float v  = X[(size_t)i * HIDDEN + c0 + c] * wv;
        atomicAdd(&acc[g * 64 + c], v);
    }
    __syncthreads();

    for (int i = threadIdx.x; i < NGRAPH * 64; i += 256) {
        const int g  = i >> 6;
        const int cc = i & 63;
        atomicAdd(&out[(size_t)g * HIDDEN + c0 + cc], acc[i]);
    }
}

// ---------------------------------------------------------------- launch
extern "C" void kernel_launch(void* const* d_in, const int* in_sizes, int n_in,
                              void* d_out, int out_size, void* d_ws, size_t ws_size,
                              hipStream_t stream) {
    const float* X  = (const float*)d_in[0];
    const float* W1 = (const float*)d_in[1];
    const float* b1 = (const float*)d_in[2];
    const float* W2 = (const float*)d_in[3];
    const float* b2 = (const float*)d_in[4];
    const int*   bi = (const int*)d_in[5];
    const int N = in_sizes[0] / HIDDEN;

    float* out = (float*)d_out;
    uint8_t* ws = (uint8_t*)d_ws;
    uint32_t* W1p    = (uint32_t*)ws;                                   // 1 MB
    float*    scores = (float*)(ws + (1u << 20));                       // N * 4 B
    unsigned* maxkey = (unsigned*)(ws + (1u << 20) + (size_t)N * 4);
    float*    sumv   = (float*)(ws + (1u << 20) + (size_t)N * 4 + 4);

    init_kernel<<<(NGRAPH * HIDDEN + 255) / 256, 256, 0, stream>>>(out, NGRAPH * HIDDEN, maxkey, sumv);
    repack_w1<<<(HIDDEN * HALF / 2 + 255) / 256, 256, 0, stream>>>(W1, W1p);

    const int nWaves = N / 16;                       // 6250
    scores_kernel<<<(nWaves + 1) / 2, 64, 0, stream>>>(X, W1p, b1, W2, b2, scores, N);

    reduce_max<<<128, 256, 0, stream>>>(scores, N, maxkey);
    reduce_sumexp<<<128, 256, 0, stream>>>(scores, N, maxkey, sumv);
    normalize_weights<<<(N + 255) / 256, 256, 0, stream>>>(scores, N, maxkey, sumv);

    scatter_kernel<<<16 * 32, 256, 0, stream>>>(X, scores, bi, out, N, 32);
}
